// Model_83958020702279
// MI455X (gfx1250) — matrix operand order repaired and checked by reading, not compiled, not run on hardware
//
#include <hip/hip_runtime.h>
#include <hip/hip_bf16.h>
#include <math.h>

#define DEV __device__ __forceinline__
#define PI_F 3.14159265358979f

typedef __attribute__((ext_vector_type(16))) _Float16 v16h;
typedef __attribute__((ext_vector_type(8)))  _Float16 v8h;
typedef __attribute__((ext_vector_type(8)))  float    v8f;
typedef __attribute__((ext_vector_type(4)))  int      v4i;

// Async global->LDS (CDNA5). Guarded so the file compiles on toolchains without
// the builtin (and on the host pass, where __gfx1250__ is undefined).
#if defined(__gfx1250__) && __has_builtin(__builtin_amdgcn_global_load_async_to_lds_b128) && __has_builtin(__builtin_amdgcn_s_wait_asynccnt)
#define HAS_ASYNC_LDS 1
#else
#define HAS_ASYNC_LDS 0
#endif

#if HAS_ASYNC_LDS
DEV void async_g2l_b128(const void* g, void* l) {
  __builtin_amdgcn_global_load_async_to_lds_b128(
      (__attribute__((address_space(1))) v4i*)(void*)g,
      (__attribute__((address_space(3))) v4i*)l, 0, 0);
}
DEV void async_wait0() { __builtin_amdgcn_s_wait_asynccnt(0); }
#endif

// ---------------------------------------------------------------------------
// WMMA fragment helpers (CDNA5 16x16x32 f16, wave32; A/B use contiguous k = 16*(lane/16) + element)
// ---------------------------------------------------------------------------
DEV v8f wmma16(v16h a, v16h b, v8f c) {
  return __builtin_amdgcn_wmma_f32_16x16x32_f16(false, a, false, b, (short)0, c, false, false);
}
DEV v16h cat8(v8h a, v8h b) {
  return __builtin_shufflevector(a, b, 0, 1, 2, 3, 4, 5, 6, 7, 8, 9, 10, 11, 12, 13, 14, 15);
}

// A fragment: 16x32 f16 tile, row stride ld halves ((ld*2)%16==0, base 16B aligned).
// lanes 0-15: M=lane, K {0..15}; lanes 16-31: M=lane-16, K {16..31}.
// Two aligned 16B loads per lane (ds_load_b128 from LDS).
DEV v16h frag_a(const _Float16* p, int ld) {
  int lane = threadIdx.x & 31;
  int row  = lane & 15;
  int kb   = (lane & 16) ? 16 : 0;
  const _Float16* q = p + row * ld + kb;
  return cat8(*(const v8h*)q, *(const v8h*)(q + 8));
}

// B fragment from LDS tile stored in fragment-swizzled order:
// element (k,c) of the 32x64 tile lives at slot=((c>>4)*2+(k>>4))*16+(c&15), half k&15,
// slot stride 24 halves (48B -> 16B aligned chunks, spread across banks).
DEV v16h frag_b_lds(const _Float16* Bs, int nt) {
  int lane = threadIdx.x & 31;
  int slot = (nt * 2 + (lane >> 4)) * 16 + (lane & 15);
  const _Float16* q = Bs + slot * 24;
  return cat8(*(const v8h*)q, *(const v8h*)(q + 8));
}

// B fragment from GLOBAL weights pre-packed in fragment order:
// addr = (((kk*NT + nt)*32) + lane)*16 halves -> 32B per lane, coalesced across the wave.
DEV v16h frag_b_g(const _Float16* Wp, int NT, int nt, int kk) {
  int lane = threadIdx.x & 31;
  const _Float16* q = Wp + ((((size_t)kk * NT + nt) * 32) + lane) * 16;
  return cat8(*(const v8h*)q, *(const v8h*)(q + 8));
}

// swizzled destination index for packing (k row, n col) with NT column tiles
DEV size_t sw_idx(int k, int n, int NT) {
  int kk = k >> 5, ki = k & 31, nt = n >> 4, col = n & 15;
  int lane = ((ki >> 4) << 4) | col;
  return ((((size_t)kk * NT + nt) * 32) + lane) * 16 + (ki & 15);
}

// ---------------------------------------------------------------------------
// Generic utility kernels
// ---------------------------------------------------------------------------
__global__ void k_zero(float* p, int n) {
  int i = blockIdx.x * 256 + threadIdx.x;
  if (i < n) p[i] = 0.f;
}
__global__ void k_cvt(const float* __restrict__ s, _Float16* __restrict__ d, int n) {
  int i = blockIdx.x * 256 + threadIdx.x;
  if (i < n) d[i] = (_Float16)s[i];
}
__global__ void k_add(float* __restrict__ a, const float* __restrict__ b, int n) {
  int i = blockIdx.x * 256 + threadIdx.x;
  if (i < n) a[i] += b[i];
}

// ---------------------------------------------------------------------------
// Big tiled WMMA GEMM: C[M,N] = act(A[M,K] * B[K,N] + bias)
// A,B f16 row-major (tight), C f32 row-major. biasmode: 0 none, 1 bias[row], 2 bias[col]
// 256 threads = 8 waves, block tile 64x64. A staged via async-to-LDS when the
// tile is fully in-bounds; B staged into a fragment-swizzled LDS layout.
// ---------------------------------------------------------------------------
DEV void store_tile(float* C, const float* bias, int M, int N, int row0, int col0,
                    v8f acc, int act, int biasmode) {
  int lane = threadIdx.x & 31;
  int col  = col0 + (lane & 15);
  int rb   = row0 + ((lane & 16) ? 8 : 0);
  if (col >= N) return;
#pragma unroll
  for (int r = 0; r < 8; ++r) {
    int row = rb + r;
    if (row < M) {
      float v = acc[r];
      if (biasmode == 1) v += bias[row];
      else if (biasmode == 2) v += bias[col];
      if (act == 1) v = v > 0.f ? v : 0.2f * v;
      else if (act == 2) v = fmaxf(v, 0.f);
      else if (act == 3) v = 1.f / (1.f + __expf(-v));
      C[(size_t)row * N + col] = v;
    }
  }
}

__global__ __launch_bounds__(256) void k_gemm(const _Float16* __restrict__ A,
                                              const _Float16* __restrict__ B,
                                              float* __restrict__ C,
                                              const float* __restrict__ bias,
                                              int M, int N, int K, int act, int biasmode) {
  __shared__ __align__(16) _Float16 As[64 * 40];
  __shared__ __align__(16) _Float16 Bs[16 * 192];   // 3072 halves, fragment-swizzled
  int bm = blockIdx.y, bn = blockIdx.x;
  int tid = threadIdx.x;
  int wave = tid >> 5;
  int mt  = wave & 3;
  int ntb = (wave >> 2) * 2;
  v8f acc0 = {}; v8f acc1 = {};
  for (int k0 = 0; k0 < K; k0 += 32) {
#if HAS_ASYNC_LDS
    bool fullA = (bm * 64 + 64 <= M) && (k0 + 32 <= K);
    if (fullA) {
      int r = tid >> 2, part = tid & 3;   // 256 chunks of 16B = whole 64x32 tile
      async_g2l_b128(A + (size_t)(bm * 64 + r) * K + k0 + part * 8,
                     &As[r * 40 + part * 8]);
    } else
#endif
    {
      for (int i = tid; i < 64 * 32; i += 256) {
        int r = i >> 5, c = i & 31;
        int gr = bm * 64 + r, gc = k0 + c;
        As[r * 40 + c] = (gr < M && gc < K) ? A[(size_t)gr * K + gc] : (_Float16)0.f;
      }
    }
    for (int i = tid; i < 32 * 64; i += 256) {
      int r = i >> 6, c = i & 63;
      int gr = k0 + r, gc = bn * 64 + c;
      _Float16 v = (gr < K && gc < N) ? B[(size_t)gr * N + gc] : (_Float16)0.f;
      int slot = ((c >> 4) * 2 + (r >> 4)) * 16 + (c & 15);
      Bs[slot * 24 + (r & 15)] = v;
    }
    if (k0 + 32 < K) {   // gfx1250 global_prefetch_b8 of next tiles (L2-resident weights)
      __builtin_prefetch(A + (size_t)(bm * 64) * K + (k0 + 32), 0, 1);
      __builtin_prefetch(B + (size_t)(k0 + 32) * N + bn * 64, 0, 1);
    }
#if HAS_ASYNC_LDS
    if (fullA) async_wait0();
#endif
    __syncthreads();
    v16h af  = frag_a(&As[(mt * 16) * 40], 40);
    v16h bf0 = frag_b_lds(Bs, ntb);
    v16h bf1 = frag_b_lds(Bs, ntb + 1);
    acc0 = wmma16(af, bf0, acc0);
    acc1 = wmma16(af, bf1, acc1);
    __syncthreads();
  }
  store_tile(C, bias, M, N, bm * 64 + mt * 16, bn * 64 + ntb * 16, acc0, act, biasmode);
  store_tile(C, bias, M, N, bm * 64 + mt * 16, bn * 64 + (ntb + 1) * 16, acc1, act, biasmode);
}

// ---------------------------------------------------------------------------
// im2col: src f32 feature layout [Cin][Bt*Lin] -> dst f16 [Cin*ks][Bt*Lout]
// repeat==1: logical input is nearest-repeat x2 along time
// ---------------------------------------------------------------------------
__global__ void k_im2col(const float* __restrict__ src, _Float16* __restrict__ dst,
                         int Cin, int Lin, int Bt, int Lout, int ks, int stride,
                         int pad_lo, int repeat) {
  int Ncol = Bt * Lout;
  int total = Cin * ks * Ncol;
  int idx = blockIdx.x * 256 + threadIdx.x;
  if (idx >= total) return;
  int n  = idx % Ncol;
  int kk = idx / Ncol;
  int ci = kk / ks, k = kk - ci * ks;
  int b = n / Lout, t = n - b * Lout;
  int pos = t * stride + k - pad_lo;
  int Llog = repeat ? Lin * 2 : Lin;
  float v = 0.f;
  if (pos >= 0 && pos < Llog) {
    int sp = repeat ? (pos >> 1) : pos;
    v = src[(size_t)ci * (Bt * Lin) + b * Lin + sp];
  }
  dst[idx] = (_Float16)v;
}

// ---------------------------------------------------------------------------
// STFT (2048-pt window, hop 256) of x (16,32768) -> spec feat layout [bin][b*128+f]
// window staged via async-to-LDS when fully in range
// ---------------------------------------------------------------------------
__global__ __launch_bounds__(256) void k_stft2048(const float* __restrict__ x,
                                                  float* __restrict__ spec) {
  int f = blockIdx.x, b = blockIdx.y, tid = threadIdx.x;
  __shared__ __align__(16) float win[2048];
  const float* xrow = x + (size_t)b * 32768;
#if HAS_ASYNC_LDS
  if (f * 256 + 2048 <= 32768) {
    for (int c = tid; c < 512; c += 256)          // 512 chunks of 16B
      async_g2l_b128(xrow + f * 256 + c * 4, &win[c * 4]);
    async_wait0();
  } else
#endif
  {
    for (int s = tid; s < 2048; s += 256) {
      int pos = f * 256 + s;
      win[s] = (pos < 32768) ? xrow[pos] : 0.f;
    }
  }
  __syncthreads();
  for (int s = tid; s < 2048; s += 256)
    win[s] *= (0.5f - 0.5f * __cosf(2.f * PI_F * s / 2048.f));
  __syncthreads();
  for (int k = tid; k < 1024; k += 256) {
    float dsn, dc;
    __sincosf(-2.f * PI_F * k / 2048.f, &dsn, &dc);
    float cc = 1.f, ss = 0.f, re = 0.f, im = 0.f;
    for (int s = 0; s < 2048; ++s) {
      re += win[s] * cc; im += win[s] * ss;
      float nc = cc * dc - ss * dsn;
      ss = cc * dsn + ss * dc; cc = nc;
    }
    spec[(size_t)k * 2048 + b * 128 + f] = sqrtf(re * re + im * im);
  }
}

// STFT (512-pt, hop 256) of impulse rows (256 x 4096, zero-extended) -> mag (n*128+f, 257)
__global__ __launch_bounds__(256) void k_stft512(const float* __restrict__ imp,
                                                 float* __restrict__ mag) {
  int f = blockIdx.x, n = blockIdx.y, tid = threadIdx.x;
  __shared__ float win[512];
  for (int s = tid; s < 512; s += 256) {
    int pos = f * 256 + s;
    float v = (pos < 4096) ? imp[(size_t)n * 4096 + pos] : 0.f;
    win[s] = v * (0.5f - 0.5f * __cosf(2.f * PI_F * s / 512.f));
  }
  __syncthreads();
  for (int k = tid; k < 257; k += 256) {
    float dsn, dc;
    __sincosf(-2.f * PI_F * k / 512.f, &dsn, &dc);
    float cc = 1.f, ss = 0.f, re = 0.f, im = 0.f;
    for (int s = 0; s < 512; ++s) {
      re += win[s] * cc; im += win[s] * ss;
      float nc = cc * dc - ss * dsn;
      ss = cc * dsn + ss * dc; cc = nc;
    }
    mag[((size_t)n * 128 + f) * 257 + k] = sqrtf(re * re + im * im);
  }
}

// ---------------------------------------------------------------------------
// Small dense helpers
// ---------------------------------------------------------------------------
__global__ void k_linear_rows(const float* __restrict__ in, const float* __restrict__ w,
                              const float* __restrict__ bias, float* __restrict__ out,
                              int R, int I, int O, int act) {
  int idx = blockIdx.x * 256 + threadIdx.x;
  if (idx >= R * O) return;
  int r = idx / O, o = idx - r * O;
  float s = bias ? bias[o] : 0.f;
  for (int i = 0; i < I; ++i) s += in[(size_t)r * I + i] * w[(size_t)i * O + o];
  if (act == 1) s = s > 0.f ? s : 0.2f * s;
  else if (act == 2) s = fmaxf(s, 0.f);
  else if (act == 3) s = 1.f / (1.f + __expf(-s));
  else if (act == 4) s = s * s;
  else if (act == 5) s = fabsf(s);
  out[idx] = s;
}

__global__ void k_mean_feat(const float* __restrict__ enc, float* __restrict__ df) {
  int idx = blockIdx.x * 256 + threadIdx.x;
  if (idx >= 16 * 4096) return;
  int b = idx / 4096, c = idx - b * 4096;
  float s = 0.f;
  for (int t = 0; t < 128; ++t) s += enc[(size_t)c * 2048 + b * 128 + t];
  df[idx] = s * (1.f / 128.f);
}

__global__ void k_dense(const float* mu, const float* sd, const float* eps, float* d) {
  int i = threadIdx.x;
  if (i < 256) d[i] = mu[i] + eps[i] * sd[i];
}

// top-16 per batch over encoded (feat layout [c][b*128+t]); scatter relu vals to enc_s
__global__ __launch_bounds__(256) void k_topk(const float* __restrict__ enc,
                                              float* __restrict__ enc_s,
                                              float* __restrict__ topv,
                                              int* __restrict__ topi) {
  int b = blockIdx.x, tid = threadIdx.x;
  __shared__ float sv[256]; __shared__ int si[256];
  __shared__ int ci[16];    __shared__ float cvv[16];
  for (int p = 0; p < 16; ++p) {
    float best = -1e30f; int bi = -1;
    for (int j = tid; j < 4096 * 128; j += 256) {
      int c = j >> 7, t = j & 127;
      float v = enc[(size_t)c * 2048 + b * 128 + t];
      bool used = false;
      for (int q = 0; q < p; ++q) if (ci[q] == j) used = true;
      if (!used && v > best) { best = v; bi = j; }
    }
    sv[tid] = best; si[tid] = bi;
    __syncthreads();
    for (int s = 128; s > 0; s >>= 1) {
      if (tid < s && sv[tid + s] > sv[tid]) { sv[tid] = sv[tid + s]; si[tid] = si[tid + s]; }
      __syncthreads();
    }
    if (tid == 0) { ci[p] = si[0]; cvv[p] = sv[0]; }
    __syncthreads();
  }
  if (tid < 16) {
    int j = ci[tid]; float v = cvv[tid];
    topv[b * 16 + tid] = v;
    topi[b * 16 + tid] = j;
    enc_s[(size_t)b * 524288 + j] = fmaxf(v, 0.f);
  }
}

__global__ void k_emb(const float* __restrict__ pctx, const float* __restrict__ ohw,
                      const float* __restrict__ ohb, const int* __restrict__ topi,
                      float* __restrict__ emb) {
  int idx = blockIdx.x * 256 + threadIdx.x;
  if (idx >= 256 * 256) return;
  int r = idx >> 8, f = idx & 255;
  int chan = topi[r] >> 7;
  emb[idx] = pctx[(r >> 4) * 256 + f] + ohw[(size_t)chan * 256 + f] + ohb[f];
}

__global__ void k_resh_frames(const float* __restrict__ hl, float* __restrict__ hfeat) {
  int idx = blockIdx.x * 256 + threadIdx.x;
  if (idx >= 256 * 512) return;
  int n = idx >> 9, j = idx & 511;
  int ch = j >> 2, p = j & 3;
  hfeat[(size_t)ch * 1024 + n * 4 + p] = hl[idx];
}

__global__ void k_env_resize(const float* __restrict__ e16, float* __restrict__ env) {
  int idx = blockIdx.x * 256 + threadIdx.x;
  if (idx >= 256 * 4096) return;
  int n = idx >> 12, s = idx & 4095;
  float coord = (s + 0.5f) * (16.f / 4096.f) - 0.5f;
  int i0 = (int)floorf(coord);
  float fr = coord - (float)i0;
  int a = i0 < 0 ? 0 : (i0 > 15 ? 15 : i0);
  int c = (i0 + 1) < 0 ? 0 : ((i0 + 1) > 15 ? 15 : i0 + 1);
  env[idx] = e16[n * 16 + a] * (1.f - fr) + e16[n * 16 + c] * fr;
}

// impulse: per row n, loop 256 frames: noise rfft(17) * sigmoid(h·Wns)^2 -> irfft(32),
// hann32, overlap-add hop 16, truncate 4096, * env
__global__ __launch_bounds__(256) void k_impulse(const float* __restrict__ hfeat,
                                                 const float* __restrict__ wns,
                                                 const float* __restrict__ bns,
                                                 const float* __restrict__ noise,
                                                 const float* __restrict__ env,
                                                 float* __restrict__ imp) {
  int n = blockIdx.x, tid = threadIdx.x;
  __shared__ float audio[4128];
  __shared__ float Yre[17], Yim[17];
  for (int i = tid; i < 4128; i += 256) audio[i] = 0.f;
  __syncthreads();
  for (int f = 0; f < 256; ++f) {
    if (tid < 17) {
      int j = tid;
      float xr = 0.f, xi = 0.f;
      for (int t = 0; t < 32; ++t) {
        float sn, cs; __sincosf(-2.f * PI_F * j * t / 32.f, &sn, &cs);
        float v = noise[f * 32 + t];
        xr += v * cs; xi += v * sn;
      }
      float acc = bns[j];
      for (int c = 0; c < 128; ++c) acc += hfeat[(size_t)c * 65536 + n * 256 + f] * wns[c * 17 + j];
      float sg = 1.f / (1.f + __expf(-acc));
      float ft = sg * sg;
      Yre[j] = ft * xr; Yim[j] = ft * xi;
    }
    __syncthreads();
    if (tid < 32) {
      int s = tid;
      float acc = Yre[0];
      for (int j = 1; j < 16; ++j) {
        float sn, cs; __sincosf(2.f * PI_F * j * s / 32.f, &sn, &cs);
        acc += 2.f * (Yre[j] * cs - Yim[j] * sn);
      }
      acc += Yre[16] * __cosf(PI_F * s);
      acc *= (1.f / 32.f);
      float w = 0.5f - 0.5f * __cosf(2.f * PI_F * s / 32.f);
      audio[f * 16 + s] += acc * w;
    }
    __syncthreads();
  }
  for (int s = tid; s < 4096; s += 256)
    imp[(size_t)n * 4096 + s] = audio[s] * env[(size_t)n * 4096 + s];
}

// ---------------------------------------------------------------------------
// row-wise norm / layernorm / npn-post (block per row)
// ---------------------------------------------------------------------------
__global__ void k_rownorm(const float* __restrict__ x, float* __restrict__ nrm, int C) {
  int r = blockIdx.x, tid = threadIdx.x;
  __shared__ float red[256];
  float q = 0.f;
  for (int k = tid; k < C; k += 256) { float v = x[(size_t)r * C + k]; q += v * v; }
  red[tid] = q; __syncthreads();
  for (int s = 128; s > 0; s >>= 1) { if (tid < s) red[tid] += red[tid + s]; __syncthreads(); }
  if (tid == 0) nrm[r] = sqrtf(red[0]);
}

__global__ void k_ln_lrelu(float* __restrict__ x, const float* __restrict__ g,
                           const float* __restrict__ be, int C) {
  int r = blockIdx.x, tid = threadIdx.x;
  __shared__ float r1[256], r2[256];
  __shared__ float mS, iS;
  float s = 0.f, q = 0.f;
  for (int k = tid; k < C; k += 256) { float v = x[(size_t)r * C + k]; s += v; q += v * v; }
  r1[tid] = s; r2[tid] = q; __syncthreads();
  for (int st = 128; st > 0; st >>= 1) {
    if (tid < st) { r1[tid] += r1[tid + st]; r2[tid] += r2[tid + st]; }
    __syncthreads();
  }
  if (tid == 0) {
    float m = r1[0] / C;
    float var = r2[0] / C - m * m;
    mS = m; iS = rsqrtf(var + 1e-5f);
  }
  __syncthreads();
  for (int k = tid; k < C; k += 256) {
    float v = (x[(size_t)r * C + k] - mS) * iS * g[k] + be[k];
    x[(size_t)r * C + k] = v > 0.f ? v : 0.2f * v;
  }
}

__global__ void k_npn_post(float* __restrict__ y, const float* __restrict__ nrm, int C) {
  int r = blockIdx.x, tid = threadIdx.x;
  __shared__ float red[256];
  __shared__ float scS;
  float q = 0.f;
  for (int k = tid; k < C; k += 256) { float v = fmaxf(y[(size_t)r * C + k], 0.f); q += v * v; }
  red[tid] = q; __syncthreads();
  for (int s = 128; s > 0; s >>= 1) { if (tid < s) red[tid] += red[tid + s]; __syncthreads(); }
  if (tid == 0) scS = nrm[r] / (sqrtf(red[0]) + 1e-8f);
  __syncthreads();
  for (int k = tid; k < C; k += 256)
    y[(size_t)r * C + k] = fmaxf(y[(size_t)r * C + k], 0.f) * scS;
}

// ---------------------------------------------------------------------------
// scan weight packing (fragment-swizzled global layout) + windowed irfft basis
// ---------------------------------------------------------------------------
__global__ void k_pack_w257(const float* __restrict__ w, _Float16* __restrict__ d) {
  int idx = blockIdx.x * 256 + threadIdx.x;
  if (idx >= 288 * 272) return;
  int k = idx / 272, n = idx - k * 272;
  float v = (k < 257 && n < 257) ? w[k * 257 + n] : 0.f;
  d[sw_idx(k, n, 17)] = (_Float16)v;
}
__global__ void k_pack_wspec(const float* __restrict__ w, _Float16* __restrict__ d) {
  int idx = blockIdx.x * 256 + threadIdx.x;
  if (idx >= 288 * 544) return;
  int k = idx / 544, n = idx - k * 544;
  float v = (k < 257 && n < 514) ? w[k * 514 + n] : 0.f;
  d[sw_idx(k, n, 34)] = (_Float16)v;
}
// basis rows 0..256: cos terms (irfft), rows 272..528: -sin terms; cols windowed by hann512
__global__ void k_make_basis(_Float16* __restrict__ d) {
  int idx = blockIdx.x * 256 + threadIdx.x;
  if (idx >= 544 * 512) return;
  int k = idx / 512, n = idx - k * 512;
  float v = 0.f;
  float w = 0.5f - 0.5f * __cosf(2.f * PI_F * n / 512.f);
  if (k < 257) {
    float c = (k == 0 || k == 256) ? 1.f : 2.f;
    v = c * __cosf(2.f * PI_F * k * n / 512.f) * (1.f / 512.f);
  } else if (k >= 272 && k < 529) {
    int kk = k - 272;
    if (kk >= 1 && kk <= 255) v = -2.f * __sinf(2.f * PI_F * kk * n / 512.f) * (1.f / 512.f);
  }
  d[sw_idx(k, n, 32)] = (_Float16)(v * w);
}

// ---------------------------------------------------------------------------
// Persistent scan kernel: 16 blocks x 16 rows, 128 steps all in LDS, WMMA GEMMs
// ---------------------------------------------------------------------------
struct ScanArgs {
  const _Float16* wt[4]; const float* bt[4]; const float* gt[3]; const float* bet[3];
  const _Float16* wl[4]; const float* bl[4]; const float* gl[3]; const float* bel[3];
  const _Float16* wspec; const _Float16* basis;
  const float* control; const float* shlat; float* audio;
};

DEV void s_gemm(const _Float16* Wp, int NT, int kch,
                const _Float16* h16s, float* out, int ldo) {
  int wave = threadIdx.x >> 5, lane = threadIdx.x & 31;
  for (int nt = wave; nt < NT; nt += 8) {
    v8f acc = {};
    for (int kk = 0; kk < kch; ++kk) {
      v16h a = frag_a(h16s + kk * 32, 544);
      v16h b = frag_b_g(Wp, NT, nt, kk);
      acc = wmma16(a, b, acc);
    }
    int col = nt * 16 + (lane & 15);
    int rb  = (lane & 16) ? 8 : 0;
#pragma unroll
    for (int r = 0; r < 8; ++r) out[(rb + r) * ldo + col] = acc[r];
  }
}
DEV void s_to_h16(const float* cur, _Float16* h16) {
  for (int i = threadIdx.x; i < 16 * 288; i += 256) {
    int r = i / 288, k = i - r * 288;
    h16[r * 544 + k] = (k < 257) ? (_Float16)cur[r * 272 + k] : (_Float16)0.f;
  }
}
DEV void s_ln(const float* y, const float* b, const float* g, const float* be, float* out) {
  int wave = threadIdx.x >> 5, lane = threadIdx.x & 31;
  for (int rr = 0; rr < 2; ++rr) {
    int r = wave * 2 + rr;
    float s = 0.f, q = 0.f;
    for (int k = lane; k < 257; k += 32) { float v = y[r * 272 + k] + b[k]; s += v; q += v * v; }
    for (int m = 16; m > 0; m >>= 1) { s += __shfl_xor(s, m, 32); q += __shfl_xor(q, m, 32); }
    float mean = s * (1.f / 257.f);
    float var  = q * (1.f / 257.f) - mean * mean;
    float inv  = rsqrtf(var + 1e-5f);
    for (int k = lane; k < 257; k += 32) {
      float v = (y[r * 272 + k] + b[k] - mean) * inv * g[k] + be[k];
      out[r * 272 + k] = v > 0.f ? v : 0.2f * v;
    }
  }
}
DEV void s_norm(const float* x, float* nrm) {
  int wave = threadIdx.x >> 5, lane = threadIdx.x & 31;
  for (int rr = 0; rr < 2; ++rr) {
    int r = wave * 2 + rr;
    float q = 0.f;
    for (int k = lane; k < 257; k += 32) { float v = x[r * 272 + k]; q += v * v; }
    for (int m = 16; m > 0; m >>= 1) q += __shfl_xor(q, m, 32);
    if (lane == 0) nrm[r] = sqrtf(q);
  }
}
DEV void s_npn(const float* y, const float* b, const float* nrm, float* out) {
  int wave = threadIdx.x >> 5, lane = threadIdx.x & 31;
  for (int rr = 0; rr < 2; ++rr) {
    int r = wave * 2 + rr;
    float q = 0.f;
    for (int k = lane; k < 257; k += 32) { float v = fmaxf(y[r * 272 + k] + b[k], 0.f); q += v * v; }
    for (int m = 16; m > 0; m >>= 1) q += __shfl_xor(q, m, 32);
    float sc = nrm[r] / (sqrtf(q) + 1e-8f);
    for (int k = lane; k < 257; k += 32)
      out[r * 272 + k] = fmaxf(y[r * 272 + k] + b[k], 0.f) * sc;
  }
}

__global__ __launch_bounds__(256) void k_scan(ScanArgs sa) {
  __shared__ float hid[16 * 272];
  __shared__ float ph[16 * 272];
  __shared__ float cur[16 * 272];
  __shared__ float yb[16 * 272];
  __shared__ __align__(16) float wide[16 * 544];  // spec out, AB matrix, samples (16x512)
  __shared__ __align__(16) _Float16 h16[16 * 544];
  __shared__ float nrm[16];
  int tid = threadIdx.x;
  int r0 = blockIdx.x * 16;
  for (int i = tid; i < 16 * 272; i += 256) { hid[i] = 0.f; ph[i] = 0.f; }
  __syncthreads();
  for (int t = 0; t < 128; ++t) {
    // hidden += ctrl_t
    for (int i = tid; i < 16 * 257; i += 256) {
      int r = i / 257, k = i - r * 257;
      hid[r * 272 + k] += sa.control[((size_t)(r0 + r) * 128 + t) * 257 + k];
    }
    __syncthreads();
    s_norm(hid, nrm);
    for (int i = tid; i < 16 * 257; i += 256) {
      int r = i / 257, k = i - r * 257;
      cur[r * 272 + k] = hid[r * 272 + k];
    }
    __syncthreads();
    // npn(transform)
    for (int l = 0; l < 3; ++l) {
      s_to_h16(cur, h16); __syncthreads();
      s_gemm(sa.wt[l], 17, 9, h16, yb, 272); __syncthreads();
      s_ln(yb, sa.bt[l], sa.gt[l], sa.bet[l], cur); __syncthreads();
    }
    s_to_h16(cur, h16); __syncthreads();
    s_gemm(sa.wt[3], 17, 9, h16, yb, 272); __syncthreads();
    s_npn(yb, sa.bt[3], nrm, hid); __syncthreads();
    // leakage los on hidden + shape_latent
    for (int i = tid; i < 16 * 257; i += 256) {
      int r = i / 257, k = i - r * 257;
      cur[r * 272 + k] = hid[r * 272 + k] + sa.shlat[(size_t)(r0 + r) * 257 + k];
    }
    __syncthreads();
    for (int l = 0; l < 3; ++l) {
      s_to_h16(cur, h16); __syncthreads();
      s_gemm(sa.wl[l], 17, 9, h16, yb, 272); __syncthreads();
      s_ln(yb, sa.bl[l], sa.gl[l], sa.bel[l], cur); __syncthreads();
    }
    s_to_h16(cur, h16); __syncthreads();
    s_gemm(sa.wl[3], 17, 9, h16, yb, 272); __syncthreads();
    // leaked = hidden * sigmoid(out)*0.2  (store in cur)
    for (int i = tid; i < 16 * 257; i += 256) {
      int r = i / 257, k = i - r * 257;
      float lv = (1.f / (1.f + __expf(-(yb[r * 272 + k] + sa.bl[3][k])))) * 0.2f;
      cur[r * 272 + k] = hid[r * 272 + k] * lv;
    }
    __syncthreads();
    // spec = leaked @ to_spec_w   (16 x 544 with cols 0..256 real, 257..513 imag)
    s_to_h16(cur, h16); __syncthreads();
    s_gemm(sa.wspec, 34, 9, h16, wide, 544); __syncthreads();
    // phase += gd * tanh(imag) * 0.1
    for (int i = tid; i < 16 * 257; i += 256) {
      int r = i / 257, k = i - r * 257;
      ph[r * 272 + k] += (PI_F * k / 256.f) * tanhf(wide[r * 544 + 257 + k]) * 0.1f;
    }
    __syncthreads();
    // build AB in place: cols 0..256 = real*cos(ph), cols 272..528 = real*sin(ph)
    for (int i = tid; i < 16 * 257; i += 256) {
      int r = i / 257, k = i - r * 257;
      float re = wide[r * 544 + k];
      float sn, cs; __sincosf(ph[r * 272 + k], &sn, &cs);
      wide[r * 544 + k] = re * cs;
      wide[r * 544 + 272 + k] = re * sn;
    }
    for (int i = tid; i < 16 * 15; i += 256) {
      int r = i / 15, k = 257 + i % 15;
      wide[r * 544 + k] = 0.f;
    }
    __syncthreads();
    for (int i = tid; i < 16 * 544; i += 256) h16[i] = (_Float16)wide[i];
    __syncthreads();
    // samples = AB @ basis  (windowed irfft), reuse wide[0..16*512) as output
    s_gemm(sa.basis, 32, 17, h16, wide, 512);
    __syncthreads();
    // overlap-add into global audio
    for (int i = tid; i < 16 * 512; i += 256) {
      int r = i >> 9, s = i & 511;
      unsafeAtomicAdd(&sa.audio[(size_t)(r0 + r) * 33024 + t * 256 + s], wide[r * 512 + s]);
    }
    // hidden -= leaked
    for (int i = tid; i < 16 * 257; i += 256) {
      int r = i / 257, k = i - r * 257;
      hid[r * 272 + k] -= cur[r * 272 + k];
    }
    __syncthreads();
  }
}

// ---------------------------------------------------------------------------
// post-scan: unit-norm rows * amps; sparse conv -> dry; reverb
// ---------------------------------------------------------------------------
__global__ void k_mix_norm(const float* __restrict__ audio, const float* __restrict__ amps,
                           float* __restrict__ mixed) {
  int n = blockIdx.x, tid = threadIdx.x;
  __shared__ float red[256];
  __shared__ float scS;
  float q = 0.f;
  for (int t = tid; t < 32768; t += 256) { float v = audio[(size_t)n * 33024 + t]; q += v * v; }
  red[tid] = q; __syncthreads();
  for (int s = 128; s > 0; s >>= 1) { if (tid < s) red[tid] += red[tid + s]; __syncthreads(); }
  if (tid == 0) scS = amps[n] / (sqrtf(red[0]) + 1e-8f);
  __syncthreads();
  for (int t = tid; t < 32768; t += 256)
    mixed[(size_t)n * 32768 + t] = audio[(size_t)n * 33024 + t] * scS;
}

__global__ void k_dry(const float* __restrict__ mixed, const float* __restrict__ topv,
                      const int* __restrict__ topi, float* __restrict__ dry) {
  int t = blockIdx.x * 256 + threadIdx.x;
  int b = blockIdx.y;
  float s = 0.f;
  for (int e = 0; e < 16; ++e) {
    float v = topv[b * 16 + e];
    int ti = topi[b * 16 + e] & 127;
    int sh = t - 256 * ti;
    if (sh >= 0) s += v * mixed[((size_t)b * 16 + e) * 32768 + sh];
  }
  dry[(size_t)b * 32768 + t] = s;
}

struct VerbArgs {
  const float* w[3]; const float* b[3]; const float* g[3]; const float* be[3];
  const float* ow; const float* ob; const float* wm; const float* bm;
  const float* dense; float* rooms; float* mix;
};
__global__ __launch_bounds__(256) void k_verb(VerbArgs va) {
  int tid = threadIdx.x, r = tid >> 4, o = tid & 15;
  __shared__ float cur[256], yq[256], zq[16 * 8], zm[16 * 2];
  cur[tid] = va.dense[r * 16 + o];
  __syncthreads();
  for (int l = 0; l < 3; ++l) {
    float s = va.b[l][o];
    for (int i = 0; i < 16; ++i) s += cur[r * 16 + i] * va.w[l][i * 16 + o];
    yq[tid] = s; __syncthreads();
    float m = 0.f, q = 0.f;
    for (int i = 0; i < 16; ++i) { float v = yq[r * 16 + i]; m += v; q += v * v; }
    m *= (1.f / 16.f); q = q * (1.f / 16.f) - m * m;
    float v = (yq[tid] - m) * rsqrtf(q + 1e-5f) * va.g[l][o] + va.be[l][o];
    cur[tid] = v > 0.f ? v : 0.2f * v;
    __syncthreads();
  }
  if (o < 8) { float s = va.ob[o]; for (int i = 0; i < 16; ++i) s += cur[r * 16 + i] * va.ow[i * 8 + o]; zq[r * 8 + o] = s; }
  if (o < 2) { float s = va.bm[o]; for (int i = 0; i < 16; ++i) s += va.dense[r * 16 + i] * va.wm[i * 2 + o]; zm[r * 2 + o] = s; }
  __syncthreads();
  if (o < 8) {
    float mx = -1e30f;
    for (int j = 0; j < 8; ++j) mx = fmaxf(mx, zq[r * 8 + j]);
    float ssum = 0.f;
    for (int j = 0; j < 8; ++j) ssum += __expf(zq[r * 8 + j] - mx);
    va.rooms[r * 8 + o] = __expf(zq[r * 8 + o] - mx) / ssum;
  }
  if (o < 2) {
    float mx = fmaxf(zm[r * 2], zm[r * 2 + 1]);
    float ssum = __expf(zm[r * 2] - mx) + __expf(zm[r * 2 + 1] - mx);
    va.mix[r * 2 + o] = __expf(zm[r * 2 + o] - mx) / ssum;
  }
}

__global__ void k_room(const float* __restrict__ rooms, const float* __restrict__ bank,
                       float* __restrict__ room) {
  int idx = blockIdx.x * 256 + threadIdx.x;
  if (idx >= 16 * 32768) return;
  int b = idx >> 15, t = idx & 32767;
  float s = 0.f;
  for (int j = 0; j < 8; ++j) s += rooms[b * 8 + j] * bank[(size_t)j * 32768 + t];
  room[idx] = s;
}

__global__ void k_wet(const float* __restrict__ dry, const float* __restrict__ room,
                      float* __restrict__ wet) {
  int idx = blockIdx.x * 256 + threadIdx.x;
  if (idx >= 16 * 32768) return;
  int b = idx >> 15, t = idx & 32767;
  float s = 0.f;
  const float* dr = dry + (size_t)b * 32768;
  const float* rm = room + (size_t)b * 32768;
  for (int u = 0; u <= t; ++u) s += dr[u] * rm[t - u];
  wet[idx] = s;
}

__global__ void k_final(const float* __restrict__ dry, const float* __restrict__ wet,
                        const float* __restrict__ mix, float* __restrict__ out) {
  int idx = blockIdx.x * 256 + threadIdx.x;
  if (idx >= 16 * 32768) return;
  int b = idx >> 15;
  out[idx] = dry[idx] * mix[b * 2 + 0] + wet[idx] * mix[b * 2 + 1];
}

// ---------------------------------------------------------------------------
// host side
// ---------------------------------------------------------------------------
static inline int cdiv(int a, int b) { return (a + b - 1) / b; }
static inline dim3 ggrid(int M, int N) { return dim3(cdiv(N, 64), cdiv(M, 64)); }

struct ChainP { const float* b[4]; const float* beta[3]; const float* g[3]; const float* w[4]; };
static ChainP chainP(void* const* d_in, int base) {
  ChainP c;
  for (int l = 0; l < 3; ++l) {
    c.b[l]    = (const float*)d_in[base + 4 * l + 0];
    c.beta[l] = (const float*)d_in[base + 4 * l + 1];
    c.g[l]    = (const float*)d_in[base + 4 * l + 2];
    c.w[l]    = (const float*)d_in[base + 4 * l + 3];
  }
  c.b[3] = (const float*)d_in[base + 12];
  c.w[3] = (const float*)d_in[base + 13];
  return c;
}

static void run_npn(hipStream_t st, const ChainP& cp, float* inb, float* tmpb, float* outb,
                    _Float16* fh, _Float16* wc, float* nrm, int R) {
  k_rownorm<<<R, 256, 0, st>>>(inb, nrm, 257);
  float* cur = inb; float* oth = tmpb;
  for (int l = 0; l < 3; ++l) {
    k_cvt<<<cdiv(257 * 257, 256), 256, 0, st>>>(cp.w[l], wc, 257 * 257);
    k_cvt<<<cdiv(R * 257, 256), 256, 0, st>>>(cur, fh, R * 257);
    k_gemm<<<ggrid(R, 257), 256, 0, st>>>(fh, wc, oth, cp.b[l], R, 257, 257, 0, 2);
    k_ln_lrelu<<<R, 256, 0, st>>>(oth, cp.g[l], cp.beta[l], 257);
    float* t = cur; cur = oth; oth = t;
  }
  k_cvt<<<cdiv(257 * 257, 256), 256, 0, st>>>(cp.w[3], wc, 257 * 257);
  k_cvt<<<cdiv(R * 257, 256), 256, 0, st>>>(cur, fh, R * 257);
  k_gemm<<<ggrid(R, 257), 256, 0, st>>>(fh, wc, outb, cp.b[3], R, 257, 257, 0, 2);
  k_npn_post<<<R, 256, 0, st>>>(outb, nrm, 257);
}

extern "C" void kernel_launch(void* const* d_in, const int* in_sizes, int n_in,
                              void* d_out, int out_size, void* d_ws, size_t ws_size,
                              hipStream_t stream) {
  (void)in_sizes; (void)n_in; (void)out_size; (void)ws_size;
  // Input order: x, eps, noise, then params flattened in JAX pytree (sorted-key) order.
  const float* x     = (const float*)d_in[0];
  const float* epsv  = (const float*)d_in[1];
  const float* noise = (const float*)d_in[2];
  #define P(i) ((const float*)d_in[i])

  float* out = (float*)d_out;
  const int OUT_ENC = 524288;
  const int OUT_IMP = 524288 + 8388608;

  // bump allocator over workspace
  char* wsb = (char*)d_ws; size_t off = 0;
  auto alloc = [&](size_t bytes) -> void* {
    void* p = wsb + off; off = (off + bytes + 255) & ~(size_t)255; return p;
  };
  const int BIG = 8650752;
  float*    specbuf = (float*)alloc((size_t)1024 * 2048 * 4);
  float*    encbuf  = (float*)alloc((size_t)4096 * 2048 * 4);   // aliased later as `mixed`
  float*    bufA    = (float*)alloc((size_t)BIG * 4);
  float*    bufB    = (float*)alloc((size_t)BIG * 4);
  float*    bufC    = (float*)alloc((size_t)BIG * 4);
  _Float16* fh16    = (_Float16*)alloc((size_t)25165824 * 2);   // aliased later as `audio`
  _Float16* wcvt    = (_Float16*)alloc((size_t)4194304 * 2);
  _Float16* wpk     = (_Float16*)alloc((size_t)8 * 78336 * 2);
  _Float16* wspec_p = (_Float16*)alloc((size_t)288 * 544 * 2);
  _Float16* basis_p = (_Float16*)alloc((size_t)544 * 512 * 2);
  float*    envbuf  = (float*)alloc((size_t)1048576 * 4);
  float*    dry     = (float*)alloc((size_t)524288 * 4);
  float*    room    = (float*)alloc((size_t)524288 * 4);
  float*    wet     = (float*)alloc((size_t)524288 * 4);
  float*    df      = (float*)alloc(65536 * 4);
  float*    mu      = (float*)alloc(256 * 4);
  float*    sdv     = (float*)alloc(256 * 4);
  float*    dense   = (float*)alloc(256 * 4);
  float*    pctx    = (float*)alloc(4096 * 4);
  float*    emb     = (float*)alloc(65536 * 4);
  float*    topv    = (float*)alloc(256 * 4);
  int*      topi    = (int*)alloc(256 * 4);
  float*    shapeb  = (float*)alloc(65792 * 4);
  float*    s1      = (float*)alloc(65792 * 4);
  float*    slat    = (float*)alloc(65792 * 4);
  float*    nrmb    = (float*)alloc(32768 * 4);
  float*    e16     = (float*)alloc(4096 * 4);
  float*    hl      = (float*)alloc(131072 * 4);
  float*    amps    = (float*)alloc(256 * 4);
  float*    roomsw  = (float*)alloc(128 * 4);
  float*    mixv    = (float*)alloc(32 * 4);
  float*    audio   = (float*)fh16;            // alias: fh16 dead before scan
  float*    mixed   = encbuf;                  // alias: enc dead after topk/mean

  // ---- STFT + UNet ----
  k_stft2048<<<dim3(128, 16), 256, 0, stream>>>(x, specbuf);
  // down1 (stride2): K=3072, N=16*64
  k_im2col<<<cdiv(3072 * 1024, 256), 256, 0, stream>>>(specbuf, fh16, 1024, 128, 16, 64, 3, 2, 0, 0);
  k_cvt<<<cdiv(1024 * 3072, 256), 256, 0, stream>>>(P(91), wcvt, 1024 * 3072);
  k_gemm<<<ggrid(1024, 1024), 256, 0, stream>>>(wcvt, fh16, bufA, P(90), 1024, 1024, 3072, 1, 1);
  // down2
  k_im2col<<<cdiv(3072 * 512, 256), 256, 0, stream>>>(bufA, fh16, 1024, 64, 16, 32, 3, 2, 0, 0);
  k_cvt<<<cdiv(1024 * 3072, 256), 256, 0, stream>>>(P(93), wcvt, 1024 * 3072);
  k_gemm<<<ggrid(1024, 512), 256, 0, stream>>>(wcvt, fh16, bufB, P(92), 1024, 512, 3072, 1, 1);
  // down3
  k_im2col<<<cdiv(3072 * 256, 256), 256, 0, stream>>>(bufB, fh16, 1024, 32, 16, 16, 3, 2, 0, 0);
  k_cvt<<<cdiv(1024 * 3072, 256), 256, 0, stream>>>(P(95), wcvt, 1024 * 3072);
  k_gemm<<<ggrid(1024, 256), 256, 0, stream>>>(wcvt, fh16, bufC, P(94), 1024, 256, 3072, 1, 1);
  // up1 (repeat + conv) + d2
  k_im2col<<<cdiv(3072 * 512, 256), 256, 0, stream>>>(bufC, fh16, 1024, 16, 16, 32, 3, 1, 1, 1);
  k_cvt<<<cdiv(1024 * 3072, 256), 256, 0, stream>>>(P(99), wcvt, 1024 * 3072);
  k_gemm<<<ggrid(1024, 512), 256, 0, stream>>>(wcvt, fh16, bufC, P(98), 1024, 512, 3072, 1, 1);
  k_add<<<cdiv(1024 * 512, 256), 256, 0, stream>>>(bufC, bufB, 1024 * 512);
  // up2 + d1
  k_im2col<<<cdiv(3072 * 1024, 256), 256, 0, stream>>>(bufC, fh16, 1024, 32, 16, 64, 3, 1, 1, 1);
  k_cvt<<<cdiv(1024 * 3072, 256), 256, 0, stream>>>(P(101), wcvt, 1024 * 3072);
  k_gemm<<<ggrid(1024, 1024), 256, 0, stream>>>(wcvt, fh16, bufB, P(100), 1024, 1024, 3072, 1, 1);
  k_add<<<cdiv(1024 * 1024, 256), 256, 0, stream>>>(bufB, bufA, 1024 * 1024);
  // up3 + x
  k_im2col<<<cdiv(3072 * 2048, 256), 256, 0, stream>>>(bufB, fh16, 1024, 64, 16, 128, 3, 1, 1, 1);
  k_cvt<<<cdiv(1024 * 3072, 256), 256, 0, stream>>>(P(103), wcvt, 1024 * 3072);
  k_gemm<<<ggrid(1024, 2048), 256, 0, stream>>>(wcvt, fh16, bufA, P(102), 1024, 2048, 3072, 1, 1);
  k_add<<<cdiv(1024 * 2048, 256), 256, 0, stream>>>(bufA, specbuf, 1024 * 2048);
  // proj 1x1 -> encoded (4096 x 2048)
  k_cvt<<<cdiv(1024 * 2048, 256), 256, 0, stream>>>(bufA, fh16, 1024 * 2048);
  k_cvt<<<cdiv(4096 * 1024, 256), 256, 0, stream>>>(P(97), wcvt, 4096 * 1024);
  k_gemm<<<ggrid(4096, 2048), 256, 0, stream>>>(wcvt, fh16, encbuf, P(96), 4096, 2048, 1024, 0, 1);

  // ---- context + sparsify ----
  k_zero<<<cdiv(8388608, 256), 256, 0, stream>>>(out + OUT_ENC, 8388608);
  k_topk<<<16, 256, 0, stream>>>(encbuf, out + OUT_ENC, topv, topi);
  k_mean_feat<<<256, 256, 0, stream>>>(encbuf, df);
  k_linear_rows<<<1, 256, 0, stream>>>(df, P(87), P(86), mu, 16, 4096, 16, 0);
  k_linear_rows<<<1, 256, 0, stream>>>(df, P(89), P(88), sdv, 16, 4096, 16, 0);
  k_dense<<<1, 256, 0, stream>>>(mu, sdv, epsv, dense);
  k_linear_rows<<<16, 256, 0, stream>>>(dense, P(6), P(5), pctx, 16, 16, 256, 0);
  k_emb<<<256, 256, 0, stream>>>(pctx, P(4), P(3), topi, emb);

  // ---- impulse generator ----
  k_linear_rows<<<16, 256, 0, stream>>>(emb, P(18), P(17), e16, 256, 256, 16, 4);
  k_env_resize<<<4096, 256, 0, stream>>>(e16, envbuf);
  k_linear_rows<<<512, 256, 0, stream>>>(emb, P(20), P(19), hl, 256, 256, 512, 0);
  k_resh_frames<<<512, 256, 0, stream>>>(hl, bufA);
  {
    const int wb[6] = {22, 24, 10, 12, 14, 16};
    const int bb[6] = {21, 23, 9, 11, 13, 15};
    float* cur = bufA; float* oth = bufC;
    int L = 4;
    for (int i = 0; i < 6; ++i) {
      int Lo = 2 * L;
      int Ncol = 256 * Lo;
      k_im2col<<<cdiv(384 * Ncol, 256), 256, 0, stream>>>(cur, fh16, 128, L, 256, Lo, 3, 1, 1, 1);
      k_cvt<<<cdiv(128 * 384, 256), 256, 0, stream>>>(P(wb[i]), wcvt, 128 * 384);
      k_gemm<<<ggrid(128, Ncol), 256, 0, stream>>>(wcvt, fh16, oth, P(bb[i]), 128, Ncol, 384, 1, 1);
      float* t = cur; cur = oth; oth = t;
      L = Lo;
    }
  }
  k_impulse<<<256, 256, 0, stream>>>(bufA, P(8), P(7), noise, envbuf, out + OUT_IMP);

  // ---- physical model ----
  k_linear_rows<<<cdiv(256 * 257, 256), 256, 0, stream>>>(emb, P(68), P(67), shapeb, 256, 256, 257, 0);
  { ChainP cs = chainP(d_in, 39); run_npn(stream, cs, shapeb, s1, slat, fh16, wcvt, nrmb, 256); }
  k_stft512<<<dim3(128, 256), 256, 0, stream>>>(out + OUT_IMP, bufA);
  { ChainP ce = chainP(d_in, 25); run_npn(stream, ce, bufA, bufC, bufB, fh16, wcvt, nrmb, 32768); }
  k_linear_rows<<<1, 256, 0, stream>>>(emb, P(85), P(84), amps, 256, 256, 1, 5);

  // pack scan weights (f16, fragment-swizzled, zero-padded 288x272) + basis
  const int trw[4] = {73, 77, 81, 83};
  const int lkw[4] = {56, 60, 64, 66};
  for (int i = 0; i < 4; ++i) {
    k_pack_w257<<<cdiv(288 * 272, 256), 256, 0, stream>>>(P(trw[i]), wpk + (size_t)i * 78336);
    k_pack_w257<<<cdiv(288 * 272, 256), 256, 0, stream>>>(P(lkw[i]), wpk + (size_t)(4 + i) * 78336);
  }
  k_pack_wspec<<<cdiv(288 * 544, 256), 256, 0, stream>>>(P(69), wspec_p);
  k_make_basis<<<cdiv(544 * 512, 256), 256, 0, stream>>>(basis_p);
  k_zero<<<cdiv(256 * 33024, 256), 256, 0, stream>>>(audio, 256 * 33024);

  ScanArgs sa;
  for (int i = 0; i < 4; ++i) { sa.wt[i] = wpk + (size_t)i * 78336; sa.wl[i] = wpk + (size_t)(4 + i) * 78336; }
  sa.bt[0] = P(70); sa.bt[1] = P(74); sa.bt[2] = P(78); sa.bt[3] = P(82);
  sa.bet[0] = P(71); sa.bet[1] = P(75); sa.bet[2] = P(79);
  sa.gt[0] = P(72); sa.gt[1] = P(76); sa.gt[2] = P(80);
  sa.bl[0] = P(53); sa.bl[1] = P(57); sa.bl[2] = P(61); sa.bl[3] = P(65);
  sa.bel[0] = P(54); sa.bel[1] = P(58); sa.bel[2] = P(62);
  sa.gl[0] = P(55); sa.gl[1] = P(59); sa.gl[2] = P(63);
  sa.wspec = wspec_p; sa.basis = basis_p;
  sa.control = bufB; sa.shlat = slat; sa.audio = audio;
  k_scan<<<16, 256, 0, stream>>>(sa);

  k_mix_norm<<<256, 256, 0, stream>>>(audio, amps, mixed);
  k_dry<<<dim3(128, 16), 256, 0, stream>>>(mixed, topv, topi, dry);

  VerbArgs va;
  va.w[0] = P(110); va.w[1] = P(114); va.w[2] = P(118);
  va.b[0] = P(107); va.b[1] = P(111); va.b[2] = P(115);
  va.be[0] = P(108); va.be[1] = P(112); va.be[2] = P(116);
  va.g[0] = P(109); va.g[1] = P(113); va.g[2] = P(117);
  va.ow = P(120); va.ob = P(119); va.wm = P(106); va.bm = P(105);
  va.dense = dense; va.rooms = roomsw; va.mix = mixv;
  k_verb<<<1, 256, 0, stream>>>(va);
  k_room<<<cdiv(16 * 32768, 256), 256, 0, stream>>>(roomsw, P(104), room);
  k_wet<<<cdiv(16 * 32768, 256), 256, 0, stream>>>(dry, room, wet);
  k_final<<<cdiv(16 * 32768, 256), 256, 0, stream>>>(dry, wet, mixv, out);
  #undef P
}
